// SelfAttention_42743514530005
// MI455X (gfx1250) — compile-verified, matrix operand order checked by reading, not run on hardware
//
#include <hip/hip_runtime.h>

// ---------------------------------------------------------------------------
// Types for CDNA5 WMMA (wave32): v_wmma_f32_16x16x32_bf16
// ---------------------------------------------------------------------------
typedef __bf16 v16bf __attribute__((ext_vector_type(16)));
typedef __bf16 v8bf  __attribute__((ext_vector_type(8)));
typedef __bf16 v4bf  __attribute__((ext_vector_type(4)));
typedef float  v8f   __attribute__((ext_vector_type(8)));

union FragBF { v16bf v; v8bf h[2]; };

__device__ __forceinline__ v8f wmma_bf16(v16bf a, v16bf b, v8f c) {
    // (neg_a, A, neg_b, B, c_mod, C, reuse_a, reuse_b)
    return __builtin_amdgcn_wmma_f32_16x16x32_bf16(false, a, false, b,
                                                   (short)0, c, false, false);
}

// Raw hardware exp2 (v_exp_f32). Inputs here are well-scaled; no range fixup.
__device__ __forceinline__ float fast_exp2(float x) {
#if __has_builtin(__builtin_amdgcn_exp2f)
    return __builtin_amdgcn_exp2f(x);
#else
    return exp2f(x);
#endif
}

// DPP16 cross-lane move (VALU pipe, no LDS). Stays within 16-lane rows.
template <int CTRL>
__device__ __forceinline__ float dpp_mov(float v) {
    return __int_as_float(__builtin_amdgcn_update_dpp(
        0, __float_as_int(v), CTRL, 0xf, 0xf, true));
}

// Reductions over each 16-lane half-wave via DPP butterflies:
//   0xB1 = quad_perm(1,0,3,2) ~ xor1 ; 0x4E = quad_perm(2,3,0,1) ~ xor2 ;
//   0x141 = row_half_mirror (^7)     ; 0x140 = row_mirror (^15)
__device__ __forceinline__ float half16_max(float v) {
    v = fmaxf(v, dpp_mov<0xB1>(v));
    v = fmaxf(v, dpp_mov<0x4E>(v));
    v = fmaxf(v, dpp_mov<0x141>(v));
    v = fmaxf(v, dpp_mov<0x140>(v));
    return v;
}
__device__ __forceinline__ float half16_sum(float v) {
    v += dpp_mov<0xB1>(v);
    v += dpp_mov<0x4E>(v);
    v += dpp_mov<0x141>(v);
    v += dpp_mov<0x140>(v);
    return v;
}

// Fragment load for 16-bit A/B matrices (16x32 tile, row stride `ld` elems).
// ISA 7.12.2: lanes 0-15 hold row M=lane, K=0..7 (lo) and K=16..23 (hi);
// lanes 16-31 hold same row, K=8..15 and K=24..31. Two ds_load_b128 per lane.
__device__ __forceinline__ v16bf ldf(const __bf16* base, int ld) {
    const int lane = threadIdx.x & 31;
    const __bf16* p = base + (lane & 15) * ld + ((lane >> 4) << 3);
    FragBF f;
    f.h[0] = *(const v8bf*)(p);
    f.h[1] = *(const v8bf*)(p + 16);
    return f.v;
}

// ---------------------------------------------------------------------------
// fp32 -> bf16 conversion (vectorized, no transpose) for activations
// ---------------------------------------------------------------------------
__global__ __launch_bounds__(256) void cvt_f32_bf16(const float* __restrict__ in,
                                                    __bf16* __restrict__ out,
                                                    int n4) {
    int i = blockIdx.x * 256 + threadIdx.x;
    if (i < n4) {
        float4 v = ((const float4*)in)[i];
        v4bf o;
        o[0] = (__bf16)v.x; o[1] = (__bf16)v.y;
        o[2] = (__bf16)v.z; o[3] = (__bf16)v.w;
        ((v4bf*)out)[i] = o;
    }
}

// ---------------------------------------------------------------------------
// fp32 -> bf16 transposing conversion for weights: out[c*R + r] = in[r*C + c]
// 32x32 LDS tile, 256 threads (32x8), both sides coalesced.
// ---------------------------------------------------------------------------
__global__ __launch_bounds__(256) void cvt_t_f32_bf16(const float* __restrict__ in,
                                                      __bf16* __restrict__ out,
                                                      int R, int C) {
    __shared__ float tile[32][33];
    const int c0 = blockIdx.x * 32, r0 = blockIdx.y * 32;
    const int tx = threadIdx.x & 31, ty = threadIdx.x >> 5;   // 32 x 8
#pragma unroll
    for (int i = 0; i < 32; i += 8)
        tile[ty + i][tx] = in[(size_t)(r0 + ty + i) * C + c0 + tx];
    __syncthreads();
#pragma unroll
    for (int i = 0; i < 32; i += 8)
        out[(size_t)(c0 + ty + i) * R + r0 + tx] = (__bf16)tile[tx][ty + i];
}

// ---------------------------------------------------------------------------
// Tiled WMMA GEMM with pre-transposed B:  C[M,N] = A[M,K] * Bt[N,K]^T
// Block 128x128, BK=64, 256 thr = 8 waves (2x4), wave tile 64x32.
// Double-buffered LDS (72 KB) + register prefetch: 1 barrier / 64-K,
// global loads of tile k+1 overlap the 16 WMMAs of tile k;
// global_prefetch of tile k+2. LDS rows padded to 72 elems -> conflict-free.
// ---------------------------------------------------------------------------
#define GLD 72
template <bool OUTF32>
__global__ __launch_bounds__(256) void gemm_bt(
    const __bf16* __restrict__ A,   // [M,K] row-major
    const __bf16* __restrict__ Bt,  // [N,K] row-major (B transposed)
    __bf16* __restrict__ Cb, float* __restrict__ Cf,
    const float* __restrict__ bias, int M, int N, int K)
{
    __shared__ __bf16 As[2][128 * GLD];
    __shared__ __bf16 Bs[2][128 * GLD];

    const int tid  = threadIdx.x;
    const int lane = tid & 31;
    const int wave = tid >> 5;
    const size_t bm = (size_t)blockIdx.y * 128;
    const size_t bn = (size_t)blockIdx.x * 128;
    const int wm = (wave >> 2) * 64;   // 2 waves along M
    const int wn = (wave & 3) * 32;    // 4 waves along N

    // staging: each thread copies 64B of A and 64B of B per tile
    const int srow = tid >> 1;            // 0..127
    const int sseg = (tid & 1) * 32;      // 0 / 32
    const __bf16* ag = A  + (bm + srow) * (size_t)K + sseg;
    const __bf16* bg = Bt + (bn + srow) * (size_t)K + sseg;

    v8f acc[4][2];
#pragma unroll
    for (int i = 0; i < 4; ++i)
#pragma unroll
        for (int j = 0; j < 2; ++j) acc[i][j] = (v8f)0.0f;

    v8bf ar[4], br[4];
    auto ldregs = [&](int k0) {
#pragma unroll
        for (int i = 0; i < 4; ++i) ar[i] = *(const v8bf*)(ag + k0 + i * 8);
#pragma unroll
        for (int i = 0; i < 4; ++i) br[i] = *(const v8bf*)(bg + k0 + i * 8);
    };
    auto stregs = [&](int buf) {
#pragma unroll
        for (int i = 0; i < 4; ++i) *(v8bf*)&As[buf][srow * GLD + sseg + i * 8] = ar[i];
#pragma unroll
        for (int i = 0; i < 4; ++i) *(v8bf*)&Bs[buf][srow * GLD + sseg + i * 8] = br[i];
    };

    ldregs(0);
    stregs(0);
    __syncthreads();

    const int nk = K >> 6;
    for (int kt = 0; kt < nk; ++kt) {
        const int cur = kt & 1;
        if (kt + 1 < nk) {
            ldregs((kt + 1) * 64);
            if (kt + 2 < nk) {
                __builtin_prefetch(ag + (kt + 2) * 64, 0, 3);
                __builtin_prefetch(bg + (kt + 2) * 64, 0, 3);
            }
        }
#pragma unroll
        for (int ks = 0; ks < 64; ks += 32) {
            v16bf af[4], bfv[2];
#pragma unroll
            for (int i = 0; i < 4; ++i) af[i]  = ldf(&As[cur][(wm + i * 16) * GLD + ks], GLD);
#pragma unroll
            for (int j = 0; j < 2; ++j) bfv[j] = ldf(&Bs[cur][(wn + j * 16) * GLD + ks], GLD);
#pragma unroll
            for (int i = 0; i < 4; ++i)
#pragma unroll
                for (int j = 0; j < 2; ++j)
                    acc[i][j] = wmma_bf16(af[i], bfv[j], acc[i][j]);
        }
        if (kt + 1 < nk) stregs(cur ^ 1);
        __syncthreads();
    }

    // epilogue (compile-time specialized, bias hoisted per column tile).
    // fp32 path = final output (write-once) -> non-temporal stores.
    const int half8 = (lane >> 4) << 3;
    const int col   = lane & 15;
#pragma unroll
    for (int j = 0; j < 2; ++j) {
        const size_t gcol = bn + wn + j * 16 + col;
        float bv = 0.0f;
        if (OUTF32) bv = bias[gcol];
#pragma unroll
        for (int i = 0; i < 4; ++i)
#pragma unroll
            for (int r = 0; r < 8; ++r) {
                const size_t grow = bm + wm + i * 16 + r + half8;
                if (OUTF32)
                    __builtin_nontemporal_store(acc[i][j][r] + bv, &Cf[grow * (size_t)N + gcol]);
                else
                    Cb[grow * (size_t)N + gcol] = (__bf16)acc[i][j][r];
            }
    }
}

// ---------------------------------------------------------------------------
// Flash attention, WMMA bf16, f32 accum, online softmax in exp2 domain
// (scale * log2(e) folded into the single post-WMMA multiply; v_exp_f32 is
// natively exp2). Row reductions via DPP16 butterflies (no LDS bpermute).
// Block = 256 thr = 8 waves; wave owns 16 q rows; block covers 128 rows of
// one (b,h). KV walked in 64-row chunks, double-buffered (1 barrier/chunk),
// chunk k+1 prefetched into registers during WMMA of chunk k.
// ---------------------------------------------------------------------------
#define KLD 72

__global__ __launch_bounds__(256) void flash_attn(
    const __bf16* __restrict__ Qb,   // [B*T, 1024]
    const __bf16* __restrict__ KVb,  // [B*T, 2048]  (k | v)
    __bf16* __restrict__ Ob)         // [B*T, 1024]  merged heads
{
    __shared__ __bf16 Klds[2][64 * KLD];   // [t][e]  == B-layout for Q·K^T
    __shared__ __bf16 Vlds[2][64 * KLD];   // [e][t]  == B-layout for P·V
    __shared__ __bf16 Plds[8 * 16 * KLD];  // wave-private P tiles (16x64)

    const int tid = threadIdx.x, lane = tid & 31, wave = tid >> 5;
    const int b = blockIdx.z, h = blockIdx.y, qt = blockIdx.x;
    const int row16 = lane & 15;
    const int kb    = (lane >> 4) << 3;    // fragment K sub-offset
    const int half8 = kb;                  // C-layout row offset (same value)
    const int q0 = qt * 128 + wave * 16;

    // Q fragments straight from global (row stride 1024 elems, 16B aligned)
    const __bf16* qp = Qb + ((size_t)(b * 2048 + q0 + row16)) * 1024 + h * 64 + kb;
    FragBF qf0, qf1;
    qf0.h[0] = *(const v8bf*)(qp);
    qf0.h[1] = *(const v8bf*)(qp + 16);
    qf1.h[0] = *(const v8bf*)(qp + 32);
    qf1.h[1] = *(const v8bf*)(qp + 48);

    float m_i[8], l_i[8];
    v8f O[4];
#pragma unroll
    for (int r = 0; r < 8; ++r) { m_i[r] = -1e30f; l_i[r] = 0.0f; }
#pragma unroll
    for (int j = 0; j < 4; ++j) O[j] = (v8f)0.0f;

    // exp2-domain scale: 1/sqrt(64) * log2(e)
    const float scale2 = 0.125f * 1.4426950408889634f;
    __bf16* Pw = &Plds[wave * 16 * KLD];

    // KV staging: each thread copies 32B of K and 32B of V per chunk
    const int trow = tid >> 2, seg = (tid & 3) * 16;
    const __bf16* kgbase = KVb + (size_t)(b * 2048 + trow) * 2048 + h * 64 + seg;

    v8bf kr0, kr1, vr0, vr1;
    auto ldregs = [&](int t0) {
        const __bf16* kg = kgbase + (size_t)t0 * 2048;
        kr0 = *(const v8bf*)(kg);
        kr1 = *(const v8bf*)(kg + 8);
        vr0 = *(const v8bf*)(kg + 1024);
        vr1 = *(const v8bf*)(kg + 1032);
    };
    auto stregs = [&](int buf) {
        *(v8bf*)&Klds[buf][trow * KLD + seg]     = kr0;
        *(v8bf*)&Klds[buf][trow * KLD + seg + 8] = kr1;
#pragma unroll
        for (int i = 0; i < 8; ++i) Vlds[buf][(seg + i) * KLD + trow]     = vr0[i];
#pragma unroll
        for (int i = 0; i < 8; ++i) Vlds[buf][(seg + 8 + i) * KLD + trow] = vr1[i];
    };

    ldregs(0);
    stregs(0);
    __syncthreads();

    for (int ci = 0; ci < 32; ++ci) {            // 32 chunks of 64 kv rows
        const int cur = ci & 1;
        if (ci + 1 < 32) {
            ldregs((ci + 1) * 64);
            if (ci + 2 < 32)
                __builtin_prefetch(kgbase + (size_t)(ci + 2) * 64 * 2048, 0, 3);
        }

        // S2 = (scale*log2e) * Q K^T  (16 x 64, 4 N-tiles, E=64 in 2 steps)
        v8f s[4];
#pragma unroll
        for (int nt = 0; nt < 4; ++nt) {
            v8f z = (v8f)0.0f;
            z = wmma_bf16(qf0.v, ldf(&Klds[cur][nt * 16 * KLD], KLD), z);
            z = wmma_bf16(qf1.v, ldf(&Klds[cur][nt * 16 * KLD + 32], KLD), z);
#pragma unroll
            for (int r = 0; r < 8; ++r) z[r] *= scale2;
            s[nt] = z;
        }

        // online softmax in exp2 domain (rows live on half-wave lane groups)
        float corr[8];
#pragma unroll
        for (int r = 0; r < 8; ++r) {
            float rm = fmaxf(fmaxf(s[0][r], s[1][r]), fmaxf(s[2][r], s[3][r]));
            rm = half16_max(rm);
            float mn = fmaxf(m_i[r], rm);
            float c  = fast_exp2(m_i[r] - mn);
            float rsum = 0.0f;
#pragma unroll
            for (int nt = 0; nt < 4; ++nt) {
                float p = fast_exp2(s[nt][r] - mn);
                s[nt][r] = p;
                rsum += p;
            }
            rsum = half16_sum(rsum);
            l_i[r] = l_i[r] * c + rsum;
            m_i[r] = mn;
            corr[r] = c;
        }
#pragma unroll
        for (int j = 0; j < 4; ++j)
#pragma unroll
            for (int r = 0; r < 8; ++r) O[j][r] *= corr[r];

        // C-layout -> A-layout reshape of P through wave-private LDS
#pragma unroll
        for (int nt = 0; nt < 4; ++nt)
#pragma unroll
            for (int r = 0; r < 8; ++r)
                Pw[(r + half8) * KLD + nt * 16 + row16] = (__bf16)s[nt][r];

        v16bf pf0 = ldf(Pw, KLD);
        v16bf pf1 = ldf(Pw + 32, KLD);

        // O += P V   (E=64 in four 16-wide tiles, Kc=64 in 2 steps)
#pragma unroll
        for (int j = 0; j < 4; ++j) {
            O[j] = wmma_bf16(pf0, ldf(&Vlds[cur][j * 16 * KLD], KLD), O[j]);
            O[j] = wmma_bf16(pf1, ldf(&Vlds[cur][j * 16 * KLD + 32], KLD), O[j]);
        }

        if (ci + 1 < 32) stregs(cur ^ 1);
        __syncthreads();
    }

    // epilogue: O / l, write merged-head bf16
#pragma unroll
    for (int r = 0; r < 8; ++r) {
        float inv = 1.0f / l_i[r];
        size_t grow = (size_t)(b * 2048 + q0 + r + half8);
#pragma unroll
        for (int j = 0; j < 4; ++j)
            Ob[grow * 1024 + h * 64 + j * 16 + row16] = (__bf16)(O[j][r] * inv);
    }
}

// ---------------------------------------------------------------------------
// Launch: cvt(x) + cvt_t(weights) -> GEMM(Q) -> GEMM(KV) -> flash_attn
//         -> GEMM(out, +bias)
// ---------------------------------------------------------------------------
extern "C" void kernel_launch(void* const* d_in, const int* in_sizes, int n_in,
                              void* d_out, int out_size, void* d_ws, size_t ws_size,
                              hipStream_t stream) {
    const int B = 4, T = 2048, D = 1024, Hh = 16, E = 64;
    const int M = B * T;            // 8192
    const int DH = Hh * E;          // 1024

    const float* x    = (const float*)d_in[0];
    const float* Wq   = (const float*)d_in[1];
    const float* Wkv  = (const float*)d_in[2];
    const float* Wout = (const float*)d_in[3];
    const float* bout = (const float*)d_in[4];
    float* out = (float*)d_out;

    char* p = (char*)d_ws;
    __bf16* Xb    = (__bf16*)p; p += (size_t)M * D * 2;        // 16 MB
    __bf16* Wqt   = (__bf16*)p; p += (size_t)D * DH * 2;       //  2 MB  [N,K]
    __bf16* Wkvt  = (__bf16*)p; p += (size_t)D * 2 * DH * 2;   //  4 MB  [2N,K]
    __bf16* Woutt = (__bf16*)p; p += (size_t)DH * D * 2;       //  2 MB  [N,K]
    __bf16* Qbuf  = (__bf16*)p; p += (size_t)M * DH * 2;       // 16 MB
    __bf16* KVbuf = (__bf16*)p; p += (size_t)M * 2 * DH * 2;   // 32 MB
    __bf16* Obuf  = (__bf16*)p;                                // 16 MB

    // 1) activations: fp32 -> bf16 ; weights: fp32 -> bf16 transposed
    {
        int n4 = (M * D) / 4;
        cvt_f32_bf16<<<(n4 + 255) / 256, 256, 0, stream>>>(x, Xb, n4);
        cvt_t_f32_bf16<<<dim3(DH / 32, D / 32),     256, 0, stream>>>(Wq,   Wqt,   D,  DH);
        cvt_t_f32_bf16<<<dim3(2 * DH / 32, D / 32), 256, 0, stream>>>(Wkv,  Wkvt,  D,  2 * DH);
        cvt_t_f32_bf16<<<dim3(D / 32, DH / 32),     256, 0, stream>>>(Wout, Woutt, DH, D);
    }

    // 2) Q = X * Wq        [8192,1024] x [1024,1024] -> bf16
    gemm_bt<false><<<dim3(DH / 128, M / 128), 256, 0, stream>>>(
        Xb, Wqt, Qbuf, nullptr, nullptr, M, DH, D);

    // 3) KV = X * Wkv      [8192,1024] x [1024,2048] -> bf16
    gemm_bt<false><<<dim3(2 * DH / 128, M / 128), 256, 0, stream>>>(
        Xb, Wkvt, KVbuf, nullptr, nullptr, M, 2 * DH, D);

    // 4) flash attention   -> Obuf (bf16, merged heads)
    flash_attn<<<dim3(T / 128, Hh, B), 256, 0, stream>>>(Qbuf, KVbuf, Obuf);

    // 5) out = Obuf * Wout + bout   -> fp32 d_out
    gemm_bt<true><<<dim3(D / 128, M / 128), 256, 0, stream>>>(
        Obuf, Woutt, nullptr, out, bout, M, D, DH);
}
